// SonNet_8967891714117
// MI455X (gfx1250) — matrix operand order repaired and checked by reading, not compiled, not run on hardware
//
#include <hip/hip_runtime.h>
#include <hip/hip_bf16.h>

// GCN on MI455X (gfx1250, wave32). bf16 WMMA GEMMs + L2-resident atomic
// scatter for the graph aggregation. All node feature matrices (25.6 MB)
// fit in the 192 MB L2; only x (410 MB) streams from HBM once.
//
// v5: ping-pong A/B operand buffers in the K loop (sets alternate and are
// redefined by loads only -> no register-rotation moves, no WMMA hazard
// NOPs); K-blocked LDS weight layout keeps all 4 column tiles in one 4 KB
// window (immediate DS offsets).

#define NFEAT 1024
#define HID   64
#define NCLASS 40

typedef __attribute__((ext_vector_type(16))) __bf16 v16bf;
typedef __attribute__((ext_vector_type(8)))  float  v8f;

// LDS layout: element index = (k>>5)*KBLK + n*32 + (k&31); KBLK = 64*32.
#define KBLK 2048

// ---- bf16 helpers (native v_cvt_pk_bf16_f32, RNE) --------------------------
static __device__ __forceinline__ __bf16 f2bf(float f) {
  return (__bf16)f;
}
static __device__ __forceinline__ unsigned short f2bfbits(float f) {
  return __builtin_bit_cast(unsigned short, (__bf16)f);
}
static __device__ __forceinline__ __bf16 bits2bf(unsigned short s) {
  return __builtin_bit_cast(__bf16, s);
}

// ---- WMMA wrapper ----------------------------------------------------------
static __device__ __forceinline__ v8f wmma_bf16(v16bf a, v16bf b, v8f c) {
  // (neg_a, A, neg_b, B, c_mod, C, reuse_a, reuse_b)
  return __builtin_amdgcn_wmma_f32_16x16x32_bf16(false, a, false, b,
                                                 (short)0, c, false, false);
}

// A operand (16x32 bf16): lane l -> row m = l&15, kh = l>>4.
// Lane holds K in {kh*16 .. kh*16+15} (contiguous, matches B LDS reads).
static __device__ __forceinline__ v16bf load_a(const float* __restrict__ rp,
                                               int kk, int kh) {
  const float* p0 = rp + kk + kh * 16;
  const float* p1 = rp + kk + kh * 16 + 8;
  float4 q0 = *(const float4*)(p0);
  float4 q1 = *(const float4*)(p0 + 4);
  float4 q2 = *(const float4*)(p1);
  float4 q3 = *(const float4*)(p1 + 4);
  v16bf a;
  a[0] = f2bf(q0.x); a[1] = f2bf(q0.y); a[2]  = f2bf(q0.z); a[3]  = f2bf(q0.w);
  a[4] = f2bf(q1.x); a[5] = f2bf(q1.y); a[6]  = f2bf(q1.z); a[7]  = f2bf(q1.w);
  a[8] = f2bf(q2.x); a[9] = f2bf(q2.y); a[10] = f2bf(q2.z); a[11] = f2bf(q2.w);
  a[12]= f2bf(q3.x); a[13]= f2bf(q3.y); a[14] = f2bf(q3.z); a[15] = f2bf(q3.w);
  return a;
}

// Fused A operand = (l1+l2+l3)/3 for the final GEMM.
static __device__ __forceinline__ v16bf load_a_avg(const float* __restrict__ r1,
                                                   const float* __restrict__ r2,
                                                   const float* __restrict__ r3,
                                                   int kk, int kh) {
  const int b0 = kk + kh * 16;
  const int b1 = kk + kh * 16 + 8;
  const float s = 1.0f / 3.0f;
  v16bf a;
#pragma unroll
  for (int j = 0; j < 8; ++j) {
    a[j]     = f2bf((r1[b0 + j] + r2[b0 + j] + r3[b0 + j]) * s);
    a[8 + j] = f2bf((r1[b1 + j] + r2[b1 + j] + r3[b1 + j]) * s);
  }
  return a;
}

// B operand (32x16 bf16): 16 contiguous shorts at the lane's slot.
static __device__ __forceinline__ v16bf ldb(const unsigned short* p) {
  v16bf b;
#pragma unroll
  for (int j = 0; j < 16; ++j) b[j] = bits2bf(p[j]);
  return b;
}

// ---- degree / norm ---------------------------------------------------------
__global__ void k_deg_init(float* __restrict__ deg, int n) {
  int i = blockIdx.x * 256 + threadIdx.x;
  if (i < n) deg[i] = 1.0f;  // self-loop
}
__global__ void k_deg_edges(const int* __restrict__ dst, float* __restrict__ deg,
                            int E) {
  int i = blockIdx.x * 256 + threadIdx.x;
  if (i < E) atomicAdd(&deg[dst[i]], 1.0f);
}
__global__ void k_dinv(const float* __restrict__ deg, float* __restrict__ dinv,
                       int n) {
  int i = blockIdx.x * 256 + threadIdx.x;
  if (i < n) dinv[i] = rsqrtf(deg[i]);
}

// ---- h = relu(x @ Wx + bx), M x 1024 @ 1024 x 64 ---------------------------
// 256 threads = 8 waves; each wave owns a 16x64 strip (4 N-tiles).
// All of Wx lives in LDS (128 KB dynamic, K-blocked layout); operand sets
// ping-pong so loads for one set overlap WMMAs of the other.
__global__ __launch_bounds__(256) void k_gemm_x(const float* __restrict__ x,
                                                const float* __restrict__ Wx,
                                                const float* __restrict__ bx,
                                                float* __restrict__ h, int M) {
  extern __shared__ unsigned short Wlds[];  // (NFEAT/32)*KBLK bf16 = 128 KB
  const int tid = threadIdx.x;
  // stage Wx: float4-coalesced global reads into the K-blocked layout
  for (int i = tid * 4; i < HID * NFEAT; i += 1024) {
    float4 w = *(const float4*)(Wx + i);
    int k = i >> 6, n = i & 63;  // 4 consecutive n, same k
    unsigned short* q = Wlds + (k >> 5) * KBLK + (k & 31);
    q[(n + 0) * 32] = f2bfbits(w.x);
    q[(n + 1) * 32] = f2bfbits(w.y);
    q[(n + 2) * 32] = f2bfbits(w.z);
    q[(n + 3) * 32] = f2bfbits(w.w);
  }
  __syncthreads();

  const int lane = tid & 31, wave = tid >> 5;
  const int kh = lane >> 4, nlo = lane & 15;
  const int m0 = blockIdx.x * 128 + wave * 16;
  int arow = m0 + nlo; if (arow >= M) arow = M - 1;
  const float* rowp = x + (size_t)arow * NFEAT;
  // lane's base slot; the 4 column tiles live at +0/+512/+1024/+1536 elems
  const unsigned short* bp = Wlds + nlo * 32 + kh * 16;

  v8f acc0 = {}, acc1 = {}, acc2 = {}, acc3 = {};
  // set 0 primed for kk=0
  v16bf a0 = load_a(rowp, 0, kh);
  v16bf b00 = ldb(bp), b01 = ldb(bp + 512), b02 = ldb(bp + 1024),
        b03 = ldb(bp + 1536);
  v16bf a1, b10, b11, b12, b13;  // set 1

  for (int kk = 0; kk < NFEAT; kk += 64) {
    // fill set 1 for kk+32 (always in range: kk <= NFEAT-64)
    a1 = load_a(rowp, kk + 32, kh);
    const unsigned short* bp1 = bp + ((kk + 32) >> 5) * KBLK;
    b10 = ldb(bp1);        b11 = ldb(bp1 + 512);
    b12 = ldb(bp1 + 1024); b13 = ldb(bp1 + 1536);
    // consume set 0
    acc0 = wmma_bf16(a0, b00, acc0);
    acc1 = wmma_bf16(a0, b01, acc1);
    acc2 = wmma_bf16(a0, b02, acc2);
    acc3 = wmma_bf16(a0, b03, acc3);
    // refill set 0 for kk+64 (skipped only on the last iteration)
    if (kk + 64 < NFEAT) {
      a0 = load_a(rowp, kk + 64, kh);
      const unsigned short* bp0 = bp + ((kk + 64) >> 5) * KBLK;
      b00 = ldb(bp0);        b01 = ldb(bp0 + 512);
      b02 = ldb(bp0 + 1024); b03 = ldb(bp0 + 1536);
    }
    // consume set 1
    acc0 = wmma_bf16(a1, b10, acc0);
    acc1 = wmma_bf16(a1, b11, acc1);
    acc2 = wmma_bf16(a1, b12, acc2);
    acc3 = wmma_bf16(a1, b13, acc3);
  }

  const int rbase = m0 + kh * 8;
  const float bb0 = bx[nlo], bb1 = bx[nlo + 16];
  const float bb2 = bx[nlo + 32], bb3 = bx[nlo + 48];
#pragma unroll
  for (int r = 0; r < 8; ++r) {
    int row = rbase + r;
    if (row < M) {
      float* o = h + (size_t)row * HID;
      o[nlo + 0]  = fmaxf(acc0[r] + bb0, 0.0f);
      o[nlo + 16] = fmaxf(acc1[r] + bb1, 0.0f);
      o[nlo + 32] = fmaxf(acc2[r] + bb2, 0.0f);
      o[nlo + 48] = fmaxf(acc3[r] + bb3, 0.0f);
    }
  }
}

// ---- per-layer GEMM: t = p @ W  and  out = t*dinv^2 + bias  (fused) --------
// K=64: both A tiles + all 8 B tiles loaded up front, 8 pipelined WMMAs.
__global__ __launch_bounds__(256) void k_gemm_h64(const float* __restrict__ p,
                                                  const float* __restrict__ W,
                                                  const float* __restrict__ dinv,
                                                  const float* __restrict__ bias,
                                                  float* __restrict__ t,
                                                  float* __restrict__ out, int M) {
  __shared__ unsigned short Wlds[2 * KBLK];  // K-blocked, 8 KB
  const int tid = threadIdx.x;
  for (int i = tid * 4; i < HID * HID; i += 1024) {
    float4 w = *(const float4*)(W + i);
    int k = i >> 6, n = i & 63;
    unsigned short* q = Wlds + (k >> 5) * KBLK + (k & 31);
    q[(n + 0) * 32] = f2bfbits(w.x);
    q[(n + 1) * 32] = f2bfbits(w.y);
    q[(n + 2) * 32] = f2bfbits(w.z);
    q[(n + 3) * 32] = f2bfbits(w.w);
  }
  __syncthreads();

  const int lane = tid & 31, wave = tid >> 5;
  const int kh = lane >> 4, nlo = lane & 15;
  const int m0 = blockIdx.x * 128 + wave * 16;
  int arow = m0 + nlo; if (arow >= M) arow = M - 1;
  const float* rowp = p + (size_t)arow * HID;
  const unsigned short* bp = Wlds + nlo * 32 + kh * 16;

  v16bf a0 = load_a(rowp, 0, kh);
  v16bf a1 = load_a(rowp, 32, kh);
  v16bf b00 = ldb(bp),        b01 = ldb(bp + 512),
        b02 = ldb(bp + 1024), b03 = ldb(bp + 1536);
  v16bf b10 = ldb(bp + KBLK),        b11 = ldb(bp + KBLK + 512),
        b12 = ldb(bp + KBLK + 1024), b13 = ldb(bp + KBLK + 1536);
  v8f acc0 = {}, acc1 = {}, acc2 = {}, acc3 = {};
  acc0 = wmma_bf16(a0, b00, acc0);
  acc1 = wmma_bf16(a0, b01, acc1);
  acc2 = wmma_bf16(a0, b02, acc2);
  acc3 = wmma_bf16(a0, b03, acc3);
  acc0 = wmma_bf16(a1, b10, acc0);
  acc1 = wmma_bf16(a1, b11, acc1);
  acc2 = wmma_bf16(a1, b12, acc2);
  acc3 = wmma_bf16(a1, b13, acc3);

  const int rbase = m0 + kh * 8;
  const float bb0 = bias[nlo], bb1 = bias[nlo + 16];
  const float bb2 = bias[nlo + 32], bb3 = bias[nlo + 48];
#pragma unroll
  for (int r = 0; r < 8; ++r) {
    int row = rbase + r;
    if (row < M) {
      float di = dinv[row];
      float d2 = di * di;
      float* ot = t + (size_t)row * HID;
      float* oo = out + (size_t)row * HID;
      float v0 = acc0[r], v1 = acc1[r], v2 = acc2[r], v3 = acc3[r];
      ot[nlo + 0]  = v0;
      ot[nlo + 16] = v1;
      ot[nlo + 32] = v2;
      ot[nlo + 48] = v3;
      oo[nlo + 0]  = v0 * d2 + bb0;   // self-loop term + bias
      oo[nlo + 16] = v1 * d2 + bb1;
      oo[nlo + 32] = v2 * d2 + bb2;
      oo[nlo + 48] = v3 * d2 + bb3;
    }
  }
}

// ---- edge scatter: one wave32 per edge, 64 feats = 2 per lane --------------
__global__ void k_scatter_edges(const float* __restrict__ t,
                                const float* __restrict__ dinv,
                                const int* __restrict__ src,
                                const int* __restrict__ dst,
                                float* __restrict__ out, int E) {
  int gw = (blockIdx.x * 256 + threadIdx.x) >> 5;
  int lane = threadIdx.x & 31;
  if (gw < E) {
    int s = src[gw], d = dst[gw];
    float norm = dinv[s] * dinv[d];
    const float* ts = t + (size_t)s * HID;
    float* od = out + (size_t)d * HID;
    atomicAdd(&od[lane],      ts[lane]      * norm);
    atomicAdd(&od[lane + 32], ts[lane + 32] * norm);
  }
}

// ---- out = relu(((l1+l2+l3)/3) @ Wz + bz), N=40 padded to 48 ---------------
__global__ __launch_bounds__(256) void k_gemm_final(const float* __restrict__ l1,
                                                    const float* __restrict__ l2,
                                                    const float* __restrict__ l3,
                                                    const float* __restrict__ Wz,
                                                    const float* __restrict__ bz,
                                                    float* __restrict__ out, int M) {
  __shared__ unsigned short Wlds[2 * KBLK];  // K-blocked, 8 KB
  const int tid = threadIdx.x;
  for (int i = tid; i < HID * HID; i += 256) {
    int k = i >> 6, n = i & 63;
    float v = (n < NCLASS) ? Wz[(size_t)k * NCLASS + n] : 0.0f;
    Wlds[(k >> 5) * KBLK + n * 32 + (k & 31)] = f2bfbits(v);
  }
  __syncthreads();
  const int lane = tid & 31, wave = tid >> 5;
  const int kh = lane >> 4, nlo = lane & 15;
  const int m0 = blockIdx.x * 128 + wave * 16;
  int arow = m0 + nlo; if (arow >= M) arow = M - 1;
  const float* r1 = l1 + (size_t)arow * HID;
  const float* r2 = l2 + (size_t)arow * HID;
  const float* r3 = l3 + (size_t)arow * HID;
  const unsigned short* bp = Wlds + nlo * 32 + kh * 16;

  v16bf a0 = load_a_avg(r1, r2, r3, 0, kh);
  v16bf a1 = load_a_avg(r1, r2, r3, 32, kh);
  v16bf b00 = ldb(bp), b01 = ldb(bp + 512), b02 = ldb(bp + 1024);
  v16bf b10 = ldb(bp + KBLK), b11 = ldb(bp + KBLK + 512),
        b12 = ldb(bp + KBLK + 1024);
  v8f acc0 = {}, acc1 = {}, acc2 = {};
  acc0 = wmma_bf16(a0, b00, acc0);
  acc1 = wmma_bf16(a0, b01, acc1);
  acc2 = wmma_bf16(a0, b02, acc2);
  acc0 = wmma_bf16(a1, b10, acc0);
  acc1 = wmma_bf16(a1, b11, acc1);
  acc2 = wmma_bf16(a1, b12, acc2);

  const int rbase = m0 + kh * 8;
#pragma unroll
  for (int r = 0; r < 8; ++r) {
    int row = rbase + r;
    if (row < M) {
      float* o = out + (size_t)row * NCLASS;
      int c0i = nlo, c1i = nlo + 16, c2i = nlo + 32;
      o[c0i] = fmaxf(acc0[r] + bz[c0i], 0.0f);
      o[c1i] = fmaxf(acc1[r] + bz[c1i], 0.0f);
      if (c2i < NCLASS) o[c2i] = fmaxf(acc2[r] + bz[c2i], 0.0f);
    }
  }
}

// ---- host-side orchestration ----------------------------------------------
extern "C" void kernel_launch(void* const* d_in, const int* in_sizes, int n_in,
                              void* d_out, int out_size, void* d_ws, size_t ws_size,
                              hipStream_t stream) {
  (void)n_in; (void)out_size; (void)ws_size;
  const float* x  = (const float*)d_in[0];
  const int*   ei = (const int*)d_in[1];
  const float* Wx = (const float*)d_in[2];
  const float* bx = (const float*)d_in[3];
  const float* W1 = (const float*)d_in[4];
  const float* b1 = (const float*)d_in[5];
  const float* W2 = (const float*)d_in[6];
  const float* b2 = (const float*)d_in[7];
  const float* W3 = (const float*)d_in[8];
  const float* b3 = (const float*)d_in[9];
  const float* Wz = (const float*)d_in[10];
  const float* bz = (const float*)d_in[11];

  const int Nn = in_sizes[0] / NFEAT;   // 100000
  const int E  = in_sizes[1] / 2;       // 1200000
  const int* src = ei;
  const int* dst = ei + E;

  // workspace carve-out
  char* ws = (char*)d_ws;
  auto take = [&](size_t bytes) {
    char* p = ws;
    ws += (bytes + 255) & ~(size_t)255;
    return p;
  };
  float* deg  = (float*)take((size_t)Nn * 4);
  float* dinv = (float*)take((size_t)Nn * 4);
  float* h    = (float*)take((size_t)Nn * HID * 4);
  float* t    = (float*)take((size_t)Nn * HID * 4);
  float* l1   = (float*)take((size_t)Nn * HID * 4);
  float* l2   = (float*)take((size_t)Nn * HID * 4);
  float* l3   = (float*)take((size_t)Nn * HID * 4);

  const dim3 B(256);
  const int gNode  = (Nn + 255) / 256;
  const int gEdge  = (E + 255) / 256;
  const int gEdgeW = (E + 7) / 8;        // 8 wave32s per block, 1 edge/wave
  const int gGemm  = (Nn + 127) / 128;   // 128 rows per block
  const size_t ldsX = (size_t)(NFEAT / 32) * KBLK * sizeof(unsigned short); // 128 KB

  // normalization
  k_deg_init <<<gNode, B, 0, stream>>>(deg, Nn);
  k_deg_edges<<<gEdge, B, 0, stream>>>(dst, deg, E);
  k_dinv     <<<gNode, B, 0, stream>>>(deg, dinv, Nn);

  // input projection (the only HBM-heavy pass)
  k_gemm_x<<<gGemm, B, ldsX, stream>>>(x, Wx, bx, h, Nn);

  // layer 1: GEMM (+fused self-loop/bias init), then edge scatter
  k_gemm_h64    <<<gGemm,  B, 0, stream>>>(h, W1, dinv, b1, t, l1, Nn);
  k_scatter_edges<<<gEdgeW, B, 0, stream>>>(t, dinv, src, dst, l1, E);
  // layer 2
  k_gemm_h64    <<<gGemm,  B, 0, stream>>>(l1, W2, dinv, b2, t, l2, Nn);
  k_scatter_edges<<<gEdgeW, B, 0, stream>>>(t, dinv, src, dst, l2, E);
  // layer 3
  k_gemm_h64    <<<gGemm,  B, 0, stream>>>(l2, W3, dinv, b3, t, l3, Nn);
  k_scatter_edges<<<gEdgeW, B, 0, stream>>>(t, dinv, src, dst, l3, E);

  // readout: fused average + GEMM + bias + relu
  k_gemm_final<<<gGemm, B, 0, stream>>>(l1, l2, l3, Wz, bz, (float*)d_out, Nn);
}
